// Decoder_Lstm_one_1236950582233
// MI455X (gfx1250) — compile-verified, matrix operand order checked by reading, not run on hardware
//
#include <hip/hip_runtime.h>
#include <hip/hip_bf16.h>

// Persistent-LSTM for MI455X (gfx1250, wave32, WMMA).
// One kernel launch per layer; grid = (ceil(H/16) column slices) x (2 batch slices).
// Each WG: stages its 64 gate rows of [Wih|Whh] into LDS as bf16 once, then loops
// 512 timesteps: stage [x_t | h_{t-1}] (fp32->bf16) into LDS, WMMA bf16 GEMM
// (f32 accum), fused sigmoid/tanh gate update with cell state held in LDS,
// write h to a double-buffered global buffer, device-scope atomic grid barrier.

typedef __attribute__((ext_vector_type(16))) __bf16 v16bf;
typedef __attribute__((ext_vector_type(8)))  float  v8f;

#define S_LEN  512
#define BATCH  256
#define RSPLIT 2
#define BR     (BATCH / RSPLIT)   // 128 batch rows per WG

__device__ __forceinline__ __bf16 f2bf(float f) {
    union { float f; unsigned u; } x; x.f = f;
    unsigned r = (x.u + 0x7FFFu + ((x.u >> 16) & 1u)) >> 16;   // round-to-nearest-even
    unsigned short s = (unsigned short)r;
    __bf16 b;
    __builtin_memcpy(&b, &s, sizeof(b));
    return b;
}

__device__ __forceinline__ float sigf(float x) { return 1.0f / (1.0f + __expf(-x)); }

// ISA 7.12.2: 16-bit A 16x32 fragment. lane half = lane>>4 selects K base (0/8),
// VGPR v<4 holds K pairs {base+2v}, v>=4 holds {base+16+2(v-4)}. B (32x16) is the
// mirrored layout with row = weight row, lane n = output column.
__device__ __forceinline__ v16bf load_frag_row(const __bf16* __restrict__ base,
                                               int stride, int row, int kc, int half) {
    v16bf f;
#pragma unroll
    for (int v = 0; v < 8; ++v) {
        int k = kc * 32 + half * 8 + 2 * v + (v >= 4 ? 8 : 0);
        f[2 * v]     = base[row * stride + k];
        f[2 * v + 1] = base[row * stride + k + 1];
    }
    return f;
}

__global__ void zero_counters_kernel(unsigned* c, int n) {
    if ((int)threadIdx.x < n) c[threadIdx.x] = 0u;
}

__global__ __launch_bounds__(256)
void lstm_layer_kernel(const float* __restrict__ xseq,  // (S, B, Din)
                       const float* __restrict__ Wih,   // (4H, Din), gate order i,f,g,o
                       const float* __restrict__ Whh,   // (4H, H)
                       const float* __restrict__ bvec,  // (4H)
                       float* __restrict__ yseq,        // (S, B, H) hidden out
                       float* __restrict__ cseq,        // (S, B, H) cell out or nullptr
                       float* __restrict__ hbuf,        // 2 * B * H fp32 double buffer
                       unsigned* __restrict__ ctr,      // per-layer grid barrier counter
                       int Din, int H, int DX32, int KC32)
{
    extern __shared__ __align__(16) char smem[];
    __bf16* lds_w  = (__bf16*)smem;                 // 64 x KC32      (weights, bf16)
    __bf16* lds_in = lds_w + 64 * KC32;             // BR x KC32      ([x|h], bf16)
    float*  lds_c  = (float*)(lds_in + BR * KC32);  // BR x 16        (cell state)

    const int tid  = threadIdx.x;
    const int lane = tid & 31;
    const int wave = tid >> 5;            // 0..7 -> 16-row batch block
    const int half = lane >> 4;           // 0/1
    const int n    = lane & 15;           // output column within tile
    const int cb   = blockIdx.x * 16;     // hidden-column base of this WG
    const int row0 = blockIdx.y * BR;     // batch-row base of this WG
    const int col  = cb + n;
    const unsigned NWGT = gridDim.x * gridDim.y;

    // Stage weight slice [Wih | Whh] -> bf16, zero padded. Rows: 4 gates x 16 cols.
    for (int idx = tid; idx < 64 * KC32; idx += 256) {
        int j = idx / KC32, k = idx - j * KC32;
        int gate = j >> 4, cj = cb + (j & 15);
        float v = 0.0f;
        if (cj < H) {
            if (k < DX32) { if (k < Din) v = Wih[(size_t)(gate * H + cj) * Din + k]; }
            else          { int kh = k - DX32; if (kh < H) v = Whh[(size_t)(gate * H + cj) * H + kh]; }
        }
        lds_w[idx] = f2bf(v);
    }
    for (int i = tid; i < BR * 16; i += 256) lds_c[i] = 0.0f;  // c0 = 0

    float bi = 0.0f, bf_ = 0.0f, bg = 0.0f, bo = 0.0f;
    if (col < H) {
        bi  = bvec[col];
        bf_ = bvec[H + col];
        bg  = bvec[2 * H + col];
        bo  = bvec[3 * H + col];
    }
    __syncthreads();

    const int KCH = KC32 >> 5;
    const int mA  = wave * 16 + n;        // A-fragment row (batch row within WG)

    for (int t = 0; t < S_LEN; ++t) {
        // ---- stage [x_t | h_{t-1}] into LDS as bf16 (zero padded) ----
        const float* xt    = xseq + (size_t)t * BATCH * Din;
        const float* hprev = hbuf + (size_t)(t & 1) * BATCH * H;
        for (int idx = tid; idx < BR * KC32; idx += 256) {
            int r = idx / KC32, k = idx - r * KC32;
            float v = 0.0f;
            if (k < DX32) {
                if (k < Din) v = xt[(size_t)(row0 + r) * Din + k];
            } else {
                int kh = k - DX32;
                if (kh < H && t > 0) v = hprev[(size_t)(row0 + r) * H + kh];
            }
            lds_in[idx] = f2bf(v);
        }
        // hint: next timestep's input slice (global_prefetch_b8)
        if (t + 1 < S_LEN) {
            const float* nx = xseq + (size_t)(t + 1) * BATCH * Din + (size_t)row0 * Din;
            __builtin_prefetch(nx + (size_t)tid * 16, 0, 1);
        }
        __syncthreads();

        // ---- gates = [x|h] @ W^T : one 16(batch)x16(col) tile per gate per wave ----
        v8f acc0 = {0,0,0,0,0,0,0,0};
        v8f acc1 = acc0, acc2 = acc0, acc3 = acc0;
        for (int kc = 0; kc < KCH; ++kc) {
            v16bf a  = load_frag_row(lds_in, KC32, mA, kc, half);
            v16bf b0 = load_frag_row(lds_w, KC32,  0 + n, kc, half);
            v16bf b1 = load_frag_row(lds_w, KC32, 16 + n, kc, half);
            v16bf b2 = load_frag_row(lds_w, KC32, 32 + n, kc, half);
            v16bf b3 = load_frag_row(lds_w, KC32, 48 + n, kc, half);
            acc0 = __builtin_amdgcn_wmma_f32_16x16x32_bf16(false, a, false, b0, (short)0, acc0, false, false);
            acc1 = __builtin_amdgcn_wmma_f32_16x16x32_bf16(false, a, false, b1, (short)0, acc1, false, false);
            acc2 = __builtin_amdgcn_wmma_f32_16x16x32_bf16(false, a, false, b2, (short)0, acc2, false, false);
            acc3 = __builtin_amdgcn_wmma_f32_16x16x32_bf16(false, a, false, b3, (short)0, acc3, false, false);
        }

        // ---- fused gate nonlinearity + cell/hidden update ----
        // D layout: VGPR r -> batch row m = wave*16 + r + 8*half, column = lane&15
#pragma unroll
        for (int r = 0; r < 8; ++r) {
            int m = wave * 16 + r + 8 * half;
            float gi = acc0[r] + bi;
            float gf = acc1[r] + bf_;
            float gg = acc2[r] + bg;
            float go = acc3[r] + bo;
            float cold = lds_c[m * 16 + n];
            float cnew = sigf(gf) * cold + sigf(gi) * tanhf(gg);
            float hnew = sigf(go) * tanhf(cnew);
            lds_c[m * 16 + n] = cnew;
            if (col < H) {
                size_t g = (size_t)(row0 + m) * H + col;
                hbuf[(size_t)((t + 1) & 1) * BATCH * H + g] = hnew;
                yseq[(size_t)t * BATCH * H + g] = hnew;
                if (cseq) cseq[(size_t)t * BATCH * H + g] = cnew;
            }
        }

        // ---- device-scope grid barrier (all WGs finished step t) ----
        __threadfence();
        __syncthreads();
        if (tid == 0) {
            __hip_atomic_fetch_add(ctr, 1u, __ATOMIC_ACQ_REL, __HIP_MEMORY_SCOPE_AGENT);
            const unsigned target = NWGT * (unsigned)(t + 1);
            while (__hip_atomic_load(ctr, __ATOMIC_ACQUIRE, __HIP_MEMORY_SCOPE_AGENT) < target) {
                __builtin_amdgcn_s_sleep(2);
            }
        }
        __syncthreads();
    }
}

extern "C" void kernel_launch(void* const* d_in, const int* in_sizes, int n_in,
                              void* d_out, int out_size, void* d_ws, size_t ws_size,
                              hipStream_t stream) {
    (void)in_sizes; (void)n_in; (void)out_size; (void)ws_size;

    const float* x = (const float*)d_in[0];

    // workspace layout (floats)
    float*    ws   = (float*)d_ws;
    unsigned* ctrs = (unsigned*)d_ws;      // 8 counters in first 64-float slot
    float* hbuf = ws + 64;                 // 2 * 256 * 256 double buffer (max H)
    const size_t SB = (size_t)S_LEN * BATCH;
    float* y0 = hbuf + 2 * 256 * 256;      // layer0 out (H=64)
    float* y1 = y0 + SB * 64;              // layer1 out (H=128)
    float* y2 = y1 + SB * 128;             // layer2 out (H=256)
    float* y3 = y2 + SB * 256;             // layer3 out (H=200)
    float* y4 = y3 + SB * 200;             // layer4 out (H=200)

    // output layout: h(l5), c(l1), c(l2), c(l3), c(l4), c(l5)
    float* out = (float*)d_out;
    float* h5 = out;
    float* c1 = h5 + SB * 200;
    float* c2 = c1 + SB * 128;
    float* c3 = c2 + SB * 256;
    float* c4 = c3 + SB * 200;
    float* c5 = c4 + SB * 200;

    zero_counters_kernel<<<dim3(1), dim3(64), 0, stream>>>(ctrs, 8);

    struct Cfg { const float* xs; float* ys; float* cs; int din, h, widx; };
    const Cfg cfg[6] = {
        { x,  y0, nullptr, 64,  64,  1  },   // bais (zero init state -> exact)
        { y0, y1, c1,      64,  128, 4  },   // l1
        { y1, y2, c2,      128, 256, 7  },   // l2
        { y2, y3, c3,      256, 200, 10 },   // l3
        { y3, y4, c4,      200, 200, 13 },   // l4
        { y4, h5, c5,      200, 200, 16 },   // l5
    };

    for (int l = 0; l < 6; ++l) {
        const int Din  = cfg[l].din, H = cfg[l].h;
        const int DX32 = (Din + 31) & ~31;
        const int H32  = (H + 31) & ~31;
        const int KC32 = DX32 + H32;
        const int ng   = (H + 15) / 16;
        const size_t lds = (size_t)(64 + BR) * KC32 * sizeof(__bf16)
                         + (size_t)BR * 16 * sizeof(float);
        const float* Wih = (const float*)d_in[cfg[l].widx];
        const float* Whh = (const float*)d_in[cfg[l].widx + 1];
        const float* bb  = (const float*)d_in[cfg[l].widx + 2];
        lstm_layer_kernel<<<dim3(ng, RSPLIT), dim3(256), lds, stream>>>(
            cfg[l].xs, Wih, Whh, bb, cfg[l].ys, cfg[l].cs, hbuf,
            ctrs + l, Din, H, DX32, KC32);
    }
}
